// AssignmentBackbone_86122684219492
// MI455X (gfx1250) — matrix operand order repaired and checked by reading, not compiled, not run on hardware
//
#include <hip/hip_runtime.h>

// ---------------------------------------------------------------------------
// Problem constants (fixed by the reference benchmark)
// ---------------------------------------------------------------------------
namespace {
constexpr int BN_  = 16;      // batch
constexpr int NN_  = 400;     // nodes per instance
constexpr int VN_  = 16;      // vehicles per instance
constexpr int KK_  = 8;       // knn degree
constexpr int Nt   = BN_ * NN_;        // 6400
constexpr int Vt   = BN_ * VN_;        // 256
constexpr int Ecnt = BN_ * VN_ * NN_;  // 102400
constexpr int EKn  = Nt * KK_;         // 51200
constexpr int HID  = 192;
constexpr int NH   = 4;
constexpr int DH   = 48;
constexpr int TDIM = 128;
}

typedef __attribute__((ext_vector_type(16))) __bf16 v16bf;
typedef __attribute__((ext_vector_type(8)))  float  v8f;

// fp32 -> packed 2x bf16 (round-to-nearest-even)
__device__ __forceinline__ unsigned int pack_bf2(float lo, float hi) {
  unsigned int a = __builtin_bit_cast(unsigned int, lo);
  unsigned int b = __builtin_bit_cast(unsigned int, hi);
  a = (a + 0x7FFFu + ((a >> 16) & 1u)) >> 16;
  b = (b + 0x7FFFu + ((b >> 16) & 1u)) & 0xFFFF0000u;
  return (a & 0xFFFFu) | b;
}

// ---------------------------------------------------------------------------
// WMMA GEMM:  C[M,N] = act(A[M,K](lda, fp32) @ W[K,N](row-major fp32) + bias)
// bf16 compute, fp32 accumulate. BM=128 BN=64 BK=32, 256 threads (8 waves).
// LDS is staged pre-swizzled so each fragment loads as 2x ds_load_b128:
//   sAu: A tile row-major bf16, row stride 20 dwords (80B, 16B aligned, no
//        bank conflicts across the 16 lanes of a fragment load)
//   sBu: W tile TRANSPOSED (row = n, 32 k-major bf16), same stride.
// ---------------------------------------------------------------------------
__global__ __launch_bounds__(256) void k_gemm(
    const float* __restrict__ A, int lda,
    const float* __restrict__ W,
    const float* __restrict__ bias,
    float* __restrict__ C, int ldc,
    int M, int Kd, int Nd, int act)
{
  __shared__ __align__(16) unsigned int sAu[128 * 20]; // 128 rows x 16 dwords (+4 pad)
  __shared__ __align__(16) unsigned int sBu[64 * 20];  // 64 n-rows x 16 dwords (+4 pad)

  const int tid  = threadIdx.x;
  const int wave = tid >> 5;
  const int lane = tid & 31;
  const int hf   = lane >> 4;      // lane half (0/1)
  const int l16  = lane & 15;
  const int m0   = blockIdx.x * 128;
  const int n0   = blockIdx.y * 64;
  const int am   = wave * 16 + l16;

  v8f acc[4] = {};

  for (int k0 = 0; k0 < Kd; k0 += 32) {
    // prefetch next K tile into the cache hierarchy (global_prefetch_b8)
    if (k0 + 32 < Kd) {
      const int pr = tid >> 4;                       // 0..15
      const int pm = (m0 + pr * 8 < M) ? (m0 + pr * 8) : (M - 1);
      __builtin_prefetch(&A[(long)pm * lda + k0 + 32 + (tid & 15) * 2], 0, 1);
      const int pk = k0 + 32 + (tid >> 6);           // 4 rows of W
      if (pk < Kd) __builtin_prefetch(&W[(long)pk * Nd + n0 + (tid & 63)], 0, 1);
    }

    // ---- stage A tile 128x32 as packed bf16 pairs --------------------------
    {
      const int dc = tid & 15;   // dword col -> k = 2*dc
      const int rb = tid >> 4;   // 16 rows per pass
      const int gk = k0 + dc * 2;
      #pragma unroll
      for (int i = 0; i < 8; ++i) {
        const int r  = rb + i * 16;
        const int gm = m0 + r;
        float lo = 0.f, hi = 0.f;
        if (gm < M) {
          if (gk < Kd)     lo = A[(long)gm * lda + gk];
          if (gk + 1 < Kd) hi = A[(long)gm * lda + gk + 1];
        }
        sAu[r * 20 + dc] = pack_bf2(lo, hi);
      }
    }
    // ---- stage W tile 32x64 transposed -------------------------------------
    {
      const int n  = tid & 63;
      const int kp = tid >> 6;   // 0..3
      const int gn = n0 + n;
      #pragma unroll
      for (int i = 0; i < 4; ++i) {
        const int dk = kp + i * 4;        // dword col -> k = 2*dk
        const int gk = k0 + dk * 2;
        float lo = 0.f, hi = 0.f;
        if (gn < Nd) {
          if (gk < Kd)     lo = W[(long)gk * Nd + gn];
          if (gk + 1 < Kd) hi = W[(long)(gk + 1) * Nd + gn];
        }
        sBu[n * 20 + dk] = pack_bf2(lo, hi);
      }
    }
    __syncthreads();

    // ---- fragments: 2x b128 each -------------------------------------------
    union { uint4 q[2]; v16bf v; } af;
    af.q[0] = *reinterpret_cast<const uint4*>(&sAu[am * 20 + hf * 4]);
    af.q[1] = *reinterpret_cast<const uint4*>(&sAu[am * 20 + 8 + hf * 4]);
    #pragma unroll
    for (int nt = 0; nt < 4; ++nt) {
      union { uint4 q[2]; v16bf v; } bfr;
      const int nrow = (nt * 16 + l16) * 20 + hf * 4;
      bfr.q[0] = *reinterpret_cast<const uint4*>(&sBu[nrow]);
      bfr.q[1] = *reinterpret_cast<const uint4*>(&sBu[nrow + 8]);
      acc[nt] = __builtin_amdgcn_wmma_f32_16x16x32_bf16(
          false, af.v, false, bfr.v, (short)0, acc[nt], false, false);
    }
    __syncthreads();
  }

  // store C (layout: VGPR r -> M = r + 8*hf, N = l16)
  const int mrow = m0 + wave * 16;
  #pragma unroll
  for (int nt = 0; nt < 4; ++nt) {
    #pragma unroll
    for (int r = 0; r < 8; ++r) {
      const int gm = mrow + r + 8 * hf;
      const int gn = n0 + nt * 16 + l16;
      if (gm < M && gn < Nd) {
        float v = acc[nt][r];
        if (bias) v += bias[gn];
        if (act == 1) v = v / (1.f + __expf(-v));   // SiLU
        C[(long)gm * ldc + gn] = v;
      }
    }
  }
}

// ---------------------------------------------------------------------------
// LayerNorm over 192 channels, wave per row.
// mode via pointers: res != 0 -> x+res ; sc != 0 -> x*(1+sc)+sh (stride scld)
// ---------------------------------------------------------------------------
__global__ void k_ln(const float* __restrict__ x, const float* __restrict__ res,
                     const float* __restrict__ sc, const float* __restrict__ sh,
                     int scld,
                     const float* __restrict__ g, const float* __restrict__ b,
                     float* __restrict__ y, int rows)
{
  const int w    = (blockIdx.x * blockDim.x + threadIdx.x) >> 5;
  const int lane = threadIdx.x & 31;
  if (w >= rows) return;
  const long base = (long)w * HID;
  float v[6];
  #pragma unroll
  for (int i = 0; i < 6; ++i) {
    const int c = lane + 32 * i;
    float t = x[base + c];
    if (res) t += res[base + c];
    if (sc)  t = t * (1.f + sc[(long)w * scld + c]) + sh[(long)w * scld + c];
    v[i] = t;
  }
  float s = v[0] + v[1] + v[2] + v[3] + v[4] + v[5];
  for (int o = 16; o; o >>= 1) s += __shfl_xor(s, o);
  const float m = s * (1.f / HID);
  float q = 0.f;
  #pragma unroll
  for (int i = 0; i < 6; ++i) { const float d = v[i] - m; q += d * d; }
  for (int o = 16; o; o >>= 1) q += __shfl_xor(q, o);
  const float r = rsqrtf(q * (1.f / HID) + 1e-5f);
  #pragma unroll
  for (int i = 0; i < 6; ++i) {
    const int c = lane + 32 * i;
    y[base + c] = (v[i] - m) * r * g[c] + b[c];
  }
}

// ---------------------------------------------------------------------------
// Attention logits: thread per (edge, head).
// logit = <q[di[e],h,:], ks[si[e],h,:] + ke[e,h,:]> / sqrt(DH) (+attr@wb+bb)
// ---------------------------------------------------------------------------
__global__ void k_logits(const float* __restrict__ q, const int* __restrict__ di,
                         const float* __restrict__ ks, const int* __restrict__ si,
                         const float* __restrict__ ke,
                         const float* __restrict__ attr,
                         const float* __restrict__ wb, const float* __restrict__ bb,
                         float* __restrict__ out, int nE)
{
  const int t = blockIdx.x * blockDim.x + threadIdx.x;
  if (t >= nE * NH) return;
  const int e = t >> 2, h = t & 3;
  const long qb = (long)di[e] * HID + h * DH;
  const long kb = (long)si[e] * HID + h * DH;
  const long eb = (long)e * HID + h * DH;
  float acc = 0.f;
  for (int d = 0; d < DH; ++d) {
    float kv = ks[kb + d];
    if (ke) kv += ke[eb + d];
    acc += q[qb + d] * kv;
  }
  acc *= 0.1443375673f;  // 1/sqrt(48)
  if (attr) {
    #pragma unroll
    for (int j = 0; j < 7; ++j) acc += attr[(long)e * 7 + j] * wb[j * NH + h];
    acc += bb[h];
  }
  out[t] = acc;
}

// Segment softmax over statically-known edge groups.
// group g -> base = (g/P)*Q + (g%P)*R ; edges = base + i*stride, i<count
__global__ void k_segsoftmax(const float* __restrict__ lg, float* __restrict__ out,
                             int groups, int count, int P, int Q, int R, int stride)
{
  const int t = blockIdx.x * blockDim.x + threadIdx.x;
  if (t >= groups * NH) return;
  const int g = t >> 2, h = t & 3;
  const long base = (long)(g / P) * Q + (long)(g % P) * R;
  float m = -1e30f;
  for (int i = 0; i < count; ++i)
    m = fmaxf(m, lg[(base + (long)i * stride) * NH + h]);
  float s = 0.f;
  for (int i = 0; i < count; ++i) {
    const long e = base + (long)i * stride;
    const float v = __expf(lg[e * NH + h] - m);
    out[e * NH + h] = v;
    s += v;
  }
  const float inv = 1.f / (s + 1e-9f);
  for (int i = 0; i < count; ++i) {
    const long e = base + (long)i * stride;
    out[e * NH + h] *= inv;
  }
}

// Weighted aggregation: out[g,c] = sum_i a[e,h(c)] * (vs[si[e],c] + ve[e,c])
__global__ void k_aggregate(const float* __restrict__ a, const int* __restrict__ si,
                            const float* __restrict__ vs, const float* __restrict__ ve,
                            float* __restrict__ out, int groups, int count,
                            int P, int Q, int R, int stride)
{
  const int t = blockIdx.x * blockDim.x + threadIdx.x;
  if (t >= groups * HID) return;
  const int g = t / HID, c = t % HID, h = c / DH;
  const long base = (long)(g / P) * Q + (long)(g % P) * R;
  float acc = 0.f;
  for (int i = 0; i < count; ++i) {
    const long e = base + (long)i * stride;
    float v = vs[(long)si[e] * HID + c];
    if (ve) v += ve[e * HID + c];
    acc += a[e * NH + h] * v;
  }
  out[(long)g * HID + c] = acc;
}

__global__ void k_mean(const float* __restrict__ x, float* __restrict__ out,
                       int nb, int count)
{
  const int t = blockIdx.x * blockDim.x + threadIdx.x;
  if (t >= nb * HID) return;
  const int b = t / HID, c = t % HID;
  float s = 0.f;
  for (int i = 0; i < count; ++i) s += x[((long)b * count + i) * HID + c];
  out[t] = s / (float)count;
}

// Scalar-output MLP tail: wave per row, out[row] (+)= <H[row,:Kd], w> + b0
__global__ void k_dot_axpy(const float* __restrict__ H, int ld, int Kd,
                           const float* __restrict__ w, const float* __restrict__ b,
                           float* __restrict__ out, int rows, int addmode)
{
  const int r    = (blockIdx.x * blockDim.x + threadIdx.x) >> 5;
  const int lane = threadIdx.x & 31;
  if (r >= rows) return;
  float acc = 0.f;
  for (int k = lane; k < Kd; k += 32) acc += H[(long)r * ld + k] * w[k];
  for (int o = 16; o; o >>= 1) acc += __shfl_xor(acc, o);
  if (lane == 0) {
    const float v = acc + b[0];
    if (addmode) out[r] += v; else out[r] = v;
  }
}

__global__ void k_add(float* __restrict__ y, const float* __restrict__ x, long n)
{
  const long i = (long)blockIdx.x * blockDim.x + threadIdx.x;
  if (i < n) y[i] += x[i];
}

__global__ void k_time_embed(const float* __restrict__ t, float* __restrict__ out)
{
  const int i = blockIdx.x * blockDim.x + threadIdx.x;
  if (i >= BN_ * TDIM) return;
  const int b = i / TDIM, c = i % TDIM;
  const int j = c & 63;
  const float f = __expf(-logf(10000.f) * (float)j / 64.f);
  const float a = t[b] * f;
  out[i] = (c < 64) ? sinf(a) : cosf(a);
}

__global__ void k_edge_in(const float* __restrict__ ea, const float* __restrict__ xt,
                          float* __restrict__ out)
{
  const int i = blockIdx.x * blockDim.x + threadIdx.x;
  if (i >= Ecnt * 5) return;
  const int e = i / 5, c = i % 5;
  out[i] = (c < 4) ? ea[(long)e * 4 + c] : xt[e];
}

__global__ void k_build_ef(const float* __restrict__ hv, const float* __restrict__ hn,
                           const float* __restrict__ hel, const float* __restrict__ edyn,
                           const float* __restrict__ xt,
                           const int* __restrict__ src, const int* __restrict__ dst,
                           float* __restrict__ ef)
{
  const int e = blockIdx.x;
  const long base = (long)e * 583;
  const int s = src[e], d = dst[e];
  for (int c = threadIdx.x; c < 583; c += blockDim.x) {
    float v;
    if      (c < 192) v = hv[(long)s * HID + c];
    else if (c < 384) v = hn[(long)d * HID + (c - 192)];
    else if (c < 576) v = hel[(long)e * HID + (c - 384)];
    else if (c < 582) v = edyn[(long)e * 6 + (c - 576)];
    else              v = xt[e];
    ef[base + c] = v;
  }
}

__global__ void k_build_feat(const float* __restrict__ heo, const float* __restrict__ hv,
                             const float* __restrict__ hn, const float* __restrict__ hg,
                             const float* __restrict__ temb, const float* __restrict__ edh,
                             const int* __restrict__ src, const int* __restrict__ dst,
                             const int* __restrict__ eg, float* __restrict__ feat)
{
  const int e = blockIdx.x;
  const long base = (long)e * 1344;
  const int s = src[e], d = dst[e], b = eg[e];
  for (int c = threadIdx.x; c < 1344; c += blockDim.x) {
    float v;
    if      (c < 192)  v = heo[(long)e * HID + c];
    else if (c < 384)  v = hv[(long)s * HID + (c - 192)];
    else if (c < 576)  v = hn[(long)d * HID + (c - 384)];
    else if (c < 768)  v = hg[(long)b * HID + (c - 576)];
    else if (c < 960)  v = temb[(long)b * HID + (c - 768)];
    else if (c < 1152) v = edh[(long)e * HID + (c - 960)];
    else               v = hv[(long)s * HID + (c - 1152)] * hn[(long)d * HID + (c - 1152)];
    feat[base + c] = v;
  }
}

__global__ void k_build_vg(const float* __restrict__ hv, const float* __restrict__ hg,
                           const float* __restrict__ temb, const int* __restrict__ vb,
                           float* __restrict__ out)
{
  const int t = blockIdx.x * blockDim.x + threadIdx.x;
  if (t >= Vt * 576) return;
  const int r = t / 576, c = t % 576;
  const int b = vb[r];
  float v;
  if      (c < 192) v = hv[(long)r * HID + c];
  else if (c < 384) v = hg[(long)b * HID + (c - 192)];
  else              v = temb[(long)b * HID + (c - 384)];
  out[t] = v;
}

__global__ void k_build_gu(const float* __restrict__ hg, const float* __restrict__ mv,
                           const float* __restrict__ mn, float* __restrict__ out)
{
  const int t = blockIdx.x * blockDim.x + threadIdx.x;
  if (t >= BN_ * 576) return;
  const int r = t / 576, c = t % 576;
  float v;
  if      (c < 192) v = hg[(long)r * HID + c];
  else if (c < 384) v = mv[(long)r * HID + (c - 192)];
  else              v = mn[(long)r * HID + (c - 384)];
  out[t] = v;
}

// ---------------------------------------------------------------------------
// Host side
// ---------------------------------------------------------------------------
static inline int cdiv(int a, int b) { return (a + b - 1) / b; }

static inline void gemm(hipStream_t s, const float* A, int lda, const float* W,
                        const float* bias, float* C, int ldc,
                        int M, int K, int N, int act)
{
  dim3 g(cdiv(M, 128), cdiv(N, 64));
  k_gemm<<<g, 256, 0, s>>>(A, lda, W, bias, C, ldc, M, K, N, act);
}

extern "C" void kernel_launch(void* const* d_in, const int* in_sizes, int n_in,
                              void* d_out, int out_size, void* d_ws, size_t ws_size,
                              hipStream_t stream)
{
  (void)in_sizes; (void)n_in; (void)out_size; (void)ws_size;

  // ---- fixed top-level inputs ------------------------------------------------
  const float* node_features = (const float*)d_in[0];
  const float* veh_features  = (const float*)d_in[1];
  const float* edge_attr     = (const float*)d_in[2];
  const float* xt01          = (const float*)d_in[3];
  const float* t_in          = (const float*)d_in[4];
  const float* edge_dyn_raw  = (const float*)d_in[5];
  const float* nn_edge_attr  = (const float*)d_in[6];
  const float* h_g_init      = (const float*)d_in[7];

  // ---- params: jax tree leaves, dict keys sorted, lists in order -------------
  int pi = 8;
  auto nx = [&]() { return (const float*)d_in[pi++]; };

  const float *ebm_b1 = nx(), *ebm_b2 = nx(), *ebm_w1 = nx(), *ebm_w2 = nx();
  const float *edb_b1[4], *edb_b2[4], *edb_w1[4], *edb_w2[4];
  for (int l = 0; l < 4; ++l) { edb_b1[l]=nx(); edb_b2[l]=nx(); edb_w1[l]=nx(); edb_w2[l]=nx(); }
  const float *edp_b1 = nx(), *edp_b2 = nx(), *edp_w1 = nx(), *edp_w2 = nx();
  const float *eh_b1 = nx(), *eh_b2 = nx(), *eh_b3 = nx(),
              *eh_w1 = nx(), *eh_w2 = nx(), *eh_w3 = nx();
  const float *emn_b[4], *emn_g[4];
  for (int l = 0; l < 4; ++l) { emn_b[l]=nx(); emn_g[l]=nx(); }
  const float *eon_b = nx(), *eon_g = nx();
  const float *ep_b1 = nx(), *ep_b2 = nx(), *ep_w1 = nx(), *ep_w2 = nx();
  const float *gp_b1 = nx(), *gp_b2 = nx(), *gp_w1 = nx(), *gp_w2 = nx();
  const float *gu_b1[4], *gu_b2[4], *gu_w1[4], *gu_w2[4];
  for (int l = 0; l < 4; ++l) { gu_b1[l]=nx(); gu_b2[l]=nx(); gu_w1[l]=nx(); gu_w2[l]=nx(); }
  const float *nn_bb[2], *nn_f_b1[2], *nn_f_b2[2], *nn_f_w1[2], *nn_f_w2[2];
  const float *nn_l1b[2], *nn_l1g[2], *nn_l2b[2], *nn_l2g[2];
  const float *nn_wb[2], *nn_wk[2], *nn_wo[2], *nn_wq[2], *nn_wv[2];
  for (int l = 0; l < 2; ++l) {
    nn_bb[l]=nx();
    nn_f_b1[l]=nx(); nn_f_b2[l]=nx(); nn_f_w1[l]=nx(); nn_f_w2[l]=nx();
    nn_l1b[l]=nx(); nn_l1g[l]=nx(); nn_l2b[l]=nx(); nn_l2g[l]=nx();
    nn_wb[l]=nx(); nn_wk[l]=nx(); nn_wo[l]=nx(); nn_wq[l]=nx(); nn_wv[l]=nx();
  }
  const float *n2v_wk[4], *n2v_wke[4], *n2v_wo[4], *n2v_wq[4], *n2v_wv[4], *n2v_wve[4];
  for (int l = 0; l < 4; ++l) {
    n2v_wk[l]=nx(); n2v_wke[l]=nx(); n2v_wo[l]=nx();
    n2v_wq[l]=nx(); n2v_wv[l]=nx(); n2v_wve[l]=nx();
  }
  const float *np_b1 = nx(), *np_b2 = nx(), *np_w1 = nx(), *np_w2 = nx();
  const float *nnc_b[4], *nnc_g[4];  for (int l=0;l<4;++l){ nnc_b[l]=nx(); nnc_g[l]=nx(); }
  const float *nni_b[4], *nni_g[4];  for (int l=0;l<4;++l){ nni_b[l]=nx(); nni_g[l]=nx(); }
  const float *nvc_b[4], *nvc_g[4];  for (int l=0;l<4;++l){ nvc_b[l]=nx(); nvc_g[l]=nx(); }
  const float *nvg_b[4], *nvg_g[4];  for (int l=0;l<4;++l){ nvg_b[l]=nx(); nvg_g[l]=nx(); }
  const float *ps_b1[3], *ps_b2[3], *ps_w1[3], *ps_w2[3];
  for (int l = 0; l < 3; ++l) { ps_b1[l]=nx(); ps_b2[l]=nx(); ps_w1[l]=nx(); ps_w2[l]=nx(); }
  const float *tp_b1 = nx(), *tp_b2 = nx(), *tp_w1 = nx(), *tp_w2 = nx();
  const float *v2n_wk[4], *v2n_wke[4], *v2n_wo[4], *v2n_wq[4], *v2n_wv[4], *v2n_wve[4];
  for (int l = 0; l < 4; ++l) {
    v2n_wk[l]=nx(); v2n_wke[l]=nx(); v2n_wo[l]=nx();
    v2n_wq[l]=nx(); v2n_wv[l]=nx(); v2n_wve[l]=nx();
  }
  const float *vp_b1 = nx(), *vp_b2 = nx(), *vp_w1 = nx(), *vp_w2 = nx();
  const float *vg_b1[4], *vg_b2[4], *vg_w1[4], *vg_w2[4];
  for (int l = 0; l < 4; ++l) { vg_b1[l]=nx(); vg_b2[l]=nx(); vg_w1[l]=nx(); vg_w2[l]=nx(); }

  const int* edge_src   = (const int*)d_in[pi++];
  const int* edge_dst   = (const int*)d_in[pi++];
  const int* edge_graph = (const int*)d_in[pi++];
  const int* node_batch = (const int*)d_in[pi++]; (void)node_batch;
  const int* veh_batch  = (const int*)d_in[pi++];
  const int* nn_src     = (const int*)d_in[pi++];
  const int* nn_dst     = (const int*)d_in[pi++];

  // ---- workspace arena (floats) ---------------------------------------------
  float* ws = (float*)d_ws;
  size_t off = 0;
  auto alloc = [&](size_t n) { size_t o = off; off += (n + 63) & ~(size_t)63; return o; };

  const size_t o_traw  = alloc((size_t)BN_ * TDIM);
  const size_t o_temb  = alloc((size_t)BN_ * HID);
  const size_t o_hg    = alloc((size_t)BN_ * HID);
  const size_t o_guout = alloc((size_t)BN_ * HID);
  const size_t o_mv    = alloc((size_t)BN_ * HID);
  const size_t o_mn    = alloc((size_t)BN_ * HID);
  const size_t o_guin  = alloc((size_t)BN_ * 576);
  const size_t o_vgin  = alloc((size_t)Vt * 576);
  const size_t o_scsh  = alloc((size_t)Vt * 384);
  const size_t o_ebias = alloc((size_t)Ecnt);
  const size_t o_hn    = alloc((size_t)Nt * HID);
  const size_t o_hv    = alloc((size_t)Vt * HID);
  const size_t o_he    = alloc((size_t)Ecnt * HID);
  const size_t o_edh   = alloc((size_t)Ecnt * HID);
  const size_t o_hel   = alloc((size_t)Ecnt * HID);
  const size_t oT      = off;                       // transient region start
  const size_t o_ke    = alloc((size_t)Ecnt * HID);
  const size_t o_ve    = alloc((size_t)Ecnt * HID);
  const size_t o_qd    = alloc((size_t)Nt * HID);
  const size_t o_ks    = alloc((size_t)Nt * HID);
  const size_t o_vs    = alloc((size_t)Nt * HID);
  const size_t o_agg   = alloc((size_t)Nt * HID);
  const size_t o_tmpA  = alloc((size_t)Nt * 384);
  const size_t o_lg    = alloc((size_t)Ecnt * NH);
  const size_t o_at    = alloc((size_t)Ecnt * NH);
  const size_t o_ef    = alloc((size_t)Ecnt * 583);
  const size_t o_ehid  = alloc((size_t)Ecnt * HID);
  // `feat` aliases the transient region (all transients are dead by then)
  const size_t o_feat  = oT;
  size_t endT = oT + (size_t)Ecnt * 1344;
  if (endT < off) endT = off;
  off = endT;
  const size_t o_x1 = alloc((size_t)Ecnt * HID);
  const size_t o_x2 = alloc((size_t)Ecnt * 96);

  const int P_BIG = 1 << 30;

  // ---- phase 0: embeddings & projections ------------------------------------
  k_time_embed<<<cdiv(BN_ * TDIM, 256), 256, 0, stream>>>(t_in, ws + o_traw);

  // h_e = mlp(edge_proj, concat(edge_attr, xt01))
  k_edge_in<<<cdiv(Ecnt * 5, 256), 256, 0, stream>>>(edge_attr, xt01, ws + o_tmpA);
  gemm(stream, ws + o_tmpA, 5, ep_w1, ep_b1, ws + o_ehid, HID, Ecnt, 5, HID, 1);
  gemm(stream, ws + o_ehid, HID, ep_w2, ep_b2, ws + o_he, HID, Ecnt, HID, HID, 0);
  // h_n
  gemm(stream, node_features, 8, np_w1, np_b1, ws + o_qd, HID, Nt, 8, HID, 1);
  gemm(stream, ws + o_qd, HID, np_w2, np_b2, ws + o_hn, HID, Nt, HID, HID, 0);
  // h_v
  gemm(stream, veh_features, 6, vp_w1, vp_b1, ws + o_ks, HID, Vt, 6, HID, 1);
  gemm(stream, ws + o_ks, HID, vp_w2, vp_b2, ws + o_hv, HID, Vt, HID, HID, 0);
  // t_emb
  gemm(stream, ws + o_traw, TDIM, tp_w1, tp_b1, ws + o_vs, HID, BN_, TDIM, HID, 1);
  gemm(stream, ws + o_vs, HID, tp_w2, tp_b2, ws + o_temb, HID, BN_, HID, HID, 0);
  // h_g
  gemm(stream, h_g_init, 8, gp_w1, gp_b1, ws + o_vs, HID, BN_, 8, HID, 1);
  gemm(stream, ws + o_vs, HID, gp_w2, gp_b2, ws + o_hg, HID, BN_, HID, HID, 0);
  // edge_dyn_h
  gemm(stream, edge_dyn_raw, 6, edp_w1, edp_b1, ws + o_ehid, HID, Ecnt, 6, HID, 1);
  gemm(stream, ws + o_ehid, HID, edp_w2, edp_b2, ws + o_edh, HID, Ecnt, HID, HID, 0);
  // edge_bias = mlp(edge_bias_mlp, edge_dyn_raw)
  gemm(stream, edge_dyn_raw, 6, ebm_w1, ebm_b1, ws + o_ehid, 96, Ecnt, 6, 96, 1);
  k_dot_axpy<<<cdiv(Ecnt * 32, 256), 256, 0, stream>>>(
      ws + o_ehid, 96, 96, ebm_w2, ebm_b2, ws + o_ebias, Ecnt, 0);

  const int refresh_idx[4] = {0, -1, 1, 2};
  const int intra_idx[4]   = {0, -1, 1, -1};

  const int lnBlkE  = cdiv(Ecnt, 8);
  const int lnBlkN  = cdiv(Nt, 8);
  const int lnBlkV  = cdiv(Vt, 8);

  for (int l = 0; l < 4; ++l) {
    // he_l = LN(edge_msg_norm[l], h_e)
    k_ln<<<lnBlkE, 256, 0, stream>>>(ws + o_he, nullptr, nullptr, nullptr, 0,
                                     emn_g[l], emn_b[l], ws + o_hel, Ecnt);

    // ---- v2n: vehicles -> nodes --------------------------------------------
    gemm(stream, ws + o_hn, HID, v2n_wq[l], nullptr, ws + o_qd, HID, Nt, HID, HID, 0);
    gemm(stream, ws + o_hv, HID, v2n_wk[l], nullptr, ws + o_ks, HID, Vt, HID, HID, 0);
    gemm(stream, ws + o_hv, HID, v2n_wv[l], nullptr, ws + o_vs, HID, Vt, HID, HID, 0);
    gemm(stream, ws + o_hel, HID, v2n_wke[l], nullptr, ws + o_ke, HID, Ecnt, HID, HID, 0);
    gemm(stream, ws + o_hel, HID, v2n_wve[l], nullptr, ws + o_ve, HID, Ecnt, HID, HID, 0);
    k_logits<<<cdiv(Ecnt * NH, 256), 256, 0, stream>>>(
        ws + o_qd, edge_dst, ws + o_ks, edge_src, ws + o_ke,
        nullptr, nullptr, nullptr, ws + o_lg, Ecnt);
    k_segsoftmax<<<cdiv(Nt * NH, 256), 256, 0, stream>>>(
        ws + o_lg, ws + o_at, Nt, VN_, NN_, VN_ * NN_, 1, NN_);
    k_aggregate<<<cdiv(Nt * HID, 256), 256, 0, stream>>>(
        ws + o_at, edge_src, ws + o_vs, ws + o_ve, ws + o_agg,
        Nt, VN_, NN_, VN_ * NN_, 1, NN_);
    gemm(stream, ws + o_agg, HID, v2n_wo[l], nullptr, ws + o_tmpA, HID, Nt, HID, HID, 0);
    k_ln<<<lnBlkN, 256, 0, stream>>>(ws + o_hn, ws + o_tmpA, nullptr, nullptr, 0,
                                     nnc_g[l], nnc_b[l], ws + o_hn, Nt);

    // ---- intra n2n (layers 0, 2) -------------------------------------------
    if (intra_idx[l] >= 0) {
      const int i2 = intra_idx[l];
      k_ln<<<lnBlkN, 256, 0, stream>>>(ws + o_hn, nullptr, nullptr, nullptr, 0,
                                       nn_l1g[i2], nn_l1b[i2], ws + o_agg, Nt);
      gemm(stream, ws + o_agg, HID, nn_wq[i2], nullptr, ws + o_qd, HID, Nt, HID, HID, 0);
      gemm(stream, ws + o_agg, HID, nn_wk[i2], nullptr, ws + o_ks, HID, Nt, HID, HID, 0);
      gemm(stream, ws + o_agg, HID, nn_wv[i2], nullptr, ws + o_vs, HID, Nt, HID, HID, 0);
      k_logits<<<cdiv(EKn * NH, 256), 256, 0, stream>>>(
          ws + o_qd, nn_dst, ws + o_ks, nn_src, nullptr,
          nn_edge_attr, nn_wb[i2], nn_bb[i2], ws + o_lg, EKn);
      k_segsoftmax<<<cdiv(Nt * NH, 256), 256, 0, stream>>>(
          ws + o_lg, ws + o_at, Nt, KK_, P_BIG, 0, KK_, 1);
      k_aggregate<<<cdiv(Nt * HID, 256), 256, 0, stream>>>(
          ws + o_at, nn_src, ws + o_vs, nullptr, ws + o_agg,
          Nt, KK_, P_BIG, 0, KK_, 1);
      gemm(stream, ws + o_agg, HID, nn_wo[i2], nullptr, ws + o_tmpA, HID, Nt, HID, HID, 0);
      k_add<<<cdiv(Nt * HID, 256), 256, 0, stream>>>(ws + o_hn, ws + o_tmpA, (long)Nt * HID);
      // FFN
      k_ln<<<lnBlkN, 256, 0, stream>>>(ws + o_hn, nullptr, nullptr, nullptr, 0,
                                       nn_l2g[i2], nn_l2b[i2], ws + o_agg, Nt);
      gemm(stream, ws + o_agg, HID, nn_f_w1[i2], nn_f_b1[i2], ws + o_tmpA, 384,
           Nt, HID, 384, 1);
      gemm(stream, ws + o_tmpA, 384, nn_f_w2[i2], nn_f_b2[i2], ws + o_agg, HID,
           Nt, 384, HID, 0);
      k_add<<<cdiv(Nt * HID, 256), 256, 0, stream>>>(ws + o_hn, ws + o_agg, (long)Nt * HID);
      k_ln<<<lnBlkN, 256, 0, stream>>>(ws + o_hn, nullptr, nullptr, nullptr, 0,
                                       nni_g[l], nni_b[l], ws + o_hn, Nt);
    }

    // ---- n2v: nodes -> vehicles (reversed edges) ----------------------------
    gemm(stream, ws + o_hv, HID, n2v_wq[l], nullptr, ws + o_qd, HID, Vt, HID, HID, 0);
    gemm(stream, ws + o_hn, HID, n2v_wk[l], nullptr, ws + o_ks, HID, Nt, HID, HID, 0);
    gemm(stream, ws + o_hn, HID, n2v_wv[l], nullptr, ws + o_vs, HID, Nt, HID, HID, 0);
    gemm(stream, ws + o_hel, HID, n2v_wke[l], nullptr, ws + o_ke, HID, Ecnt, HID, HID, 0);
    gemm(stream, ws + o_hel, HID, n2v_wve[l], nullptr, ws + o_ve, HID, Ecnt, HID, HID, 0);
    k_logits<<<cdiv(Ecnt * NH, 256), 256, 0, stream>>>(
        ws + o_qd, edge_src, ws + o_ks, edge_dst, ws + o_ke,
        nullptr, nullptr, nullptr, ws + o_lg, Ecnt);
    k_segsoftmax<<<cdiv(Vt * NH, 256), 256, 0, stream>>>(
        ws + o_lg, ws + o_at, Vt, NN_, P_BIG, 0, NN_, 1);
    k_aggregate<<<cdiv(Vt * HID, 256), 256, 0, stream>>>(
        ws + o_at, edge_dst, ws + o_vs, ws + o_ve, ws + o_agg,
        Vt, NN_, P_BIG, 0, NN_, 1);
    gemm(stream, ws + o_agg, HID, n2v_wo[l], nullptr, ws + o_tmpA, HID, Vt, HID, HID, 0);
    k_ln<<<lnBlkV, 256, 0, stream>>>(ws + o_hv, ws + o_tmpA, nullptr, nullptr, 0,
                                     nvc_g[l], nvc_b[l], ws + o_hv, Vt);

    // ---- global conditioning ------------------------------------------------
    k_mean<<<cdiv(BN_ * HID, 256), 256, 0, stream>>>(ws + o_hv, ws + o_mv, BN_, VN_);
    k_mean<<<cdiv(BN_ * HID, 256), 256, 0, stream>>>(ws + o_hn, ws + o_mn, BN_, NN_);
    k_build_vg<<<cdiv(Vt * 576, 256), 256, 0, stream>>>(
        ws + o_hv, ws + o_hg, ws + o_temb, veh_batch, ws + o_vgin);
    gemm(stream, ws + o_vgin, 576, vg_w1[l], vg_b1[l], ws + o_tmpA, HID, Vt, 576, HID, 1);
    gemm(stream, ws + o_tmpA, HID, vg_w2[l], vg_b2[l], ws + o_scsh, 384, Vt, HID, 384, 0);
    k_ln<<<lnBlkV, 256, 0, stream>>>(ws + o_hv, nullptr, ws + o_scsh, ws + o_scsh + 192,
                                     384, nvg_g[l], nvg_b[l], ws + o_hv, Vt);
    k_build_gu<<<cdiv(BN_ * 576, 256), 256, 0, stream>>>(
        ws + o_hg, ws + o_mv, ws + o_mn, ws + o_guin);
    gemm(stream, ws + o_guin, 576, gu_w1[l], gu_b1[l], ws + o_tmpA, HID, BN_, 576, HID, 1);
    gemm(stream, ws + o_tmpA, HID, gu_w2[l], gu_b2[l], ws + o_guout, HID, BN_, HID, HID, 0);
    k_add<<<cdiv(BN_ * HID, 256), 256, 0, stream>>>(ws + o_hg, ws + o_guout, (long)BN_ * HID);

    // ---- per-layer edge bias refinement ------------------------------------
    k_build_ef<<<Ecnt, 256, 0, stream>>>(ws + o_hv, ws + o_hn, ws + o_hel,
                                         edge_dyn_raw, xt01, edge_src, edge_dst,
                                         ws + o_ef);
    gemm(stream, ws + o_ef, 583, edb_w1[l], edb_b1[l], ws + o_ehid, 96, Ecnt, 583, 96, 1);
    k_dot_axpy<<<cdiv(Ecnt * 32, 256), 256, 0, stream>>>(
        ws + o_ehid, 96, 96, edb_w2[l], edb_b2[l], ws + o_ebias, Ecnt, 1);
    if (refresh_idx[l] >= 0) {
      const int r = refresh_idx[l];
      gemm(stream, ws + o_ef, 583, ps_w1[r], ps_b1[r], ws + o_ehid, HID, Ecnt, 576, HID, 1);
      k_dot_axpy<<<cdiv(Ecnt * 32, 256), 256, 0, stream>>>(
          ws + o_ehid, HID, HID, ps_w2[r], ps_b2[r], ws + o_ebias, Ecnt, 1);
    }
  }

  // ---- edge head -------------------------------------------------------------
  k_ln<<<lnBlkE, 256, 0, stream>>>(ws + o_he, nullptr, nullptr, nullptr, 0,
                                   eon_g, eon_b, ws + o_hel, Ecnt);   // he_out
  k_build_feat<<<Ecnt, 256, 0, stream>>>(ws + o_hel, ws + o_hv, ws + o_hn, ws + o_hg,
                                         ws + o_temb, ws + o_edh,
                                         edge_src, edge_dst, edge_graph, ws + o_feat);
  gemm(stream, ws + o_feat, 1344, eh_w1, eh_b1, ws + o_x1, HID, Ecnt, 1344, HID, 1);
  gemm(stream, ws + o_x1, HID, eh_w2, eh_b2, ws + o_x2, 96, Ecnt, HID, 96, 1);

  hipMemcpyAsync(d_out, ws + o_ebias, (size_t)Ecnt * sizeof(float),
                 hipMemcpyDeviceToDevice, stream);
  k_dot_axpy<<<cdiv(Ecnt * 32, 256), 256, 0, stream>>>(
      ws + o_x2, 96, 96, eh_w3, eh_b3, (float*)d_out, Ecnt, 1);
}
